// CausalSelfAttention_70497593196743
// MI455X (gfx1250) — matrix operand order repaired and checked by reading, not compiled, not run on hardware
//
#include <hip/hip_runtime.h>
#include <hip/hip_bf16.h>

// ---------------------------------------------------------------------------
// CausalSelfAttention (GQA, RMSNorm + RoPE) for MI455X / gfx1250, f16 WMMA.
// B=1, T=4096, DIM=1024, H=16, HKV=4, HD=64, RD=64, BASE=10000
// ---------------------------------------------------------------------------

#define T_LEN 4096
#define DIM   1024
#define NH    16
#define NKV   4
#define HD    64
#define QKVN  1536   // DIM + 2*HKV*HD

typedef __attribute__((ext_vector_type(16))) _Float16 v16h;
typedef __attribute__((ext_vector_type(8)))  _Float16 v8h;
typedef __attribute__((ext_vector_type(8)))  float    v8f;

// Pointer-based fragment loads: caller passes the per-lane resolved address.
// A-fragment (16x32): chunks of 8 halfs at p and p+16 (lane: row=l&15, kchunk=(l>>4)*8)
__device__ __forceinline__ v16h load_a16(const _Float16* p) {
    v8h lo = *(const v8h*)(p);
    v8h hi = *(const v8h*)(p + 16);
    v16h r;
#pragma unroll
    for (int i = 0; i < 8; ++i) { r[i] = lo[i]; r[i + 8] = hi[i]; }
    return r;
}
// B-fragment (32x16): chunks of 8 halfs at p and p+16 (lane: col=l&15, kchunk=(l>>4)*8)
__device__ __forceinline__ v16h load_b16(const _Float16* p) {
    v8h lo = *(const v8h*)(p);
    v8h hi = *(const v8h*)(p + 16);
    v16h r;
#pragma unroll
    for (int i = 0; i < 8; ++i) { r[i] = lo[i]; r[i + 8] = hi[i]; }
    return r;
}

__device__ __forceinline__ v8f wmma_f16(v16h a, v16h b, v8f c) {
    return __builtin_amdgcn_wmma_f32_16x16x32_f16(false, a, false, b, (short)0, c, false, false);
}

// ---------------------------------------------------------------------------
__global__ void cast_f32_to_f16(const float* __restrict__ in,
                                _Float16* __restrict__ out, int n) {
    int i = blockIdx.x * blockDim.x + threadIdx.x;
    if (i < n) out[i] = (_Float16)in[i];
}

// ---------------------------------------------------------------------------
// C[M,N] (f32, row-major) = A[M,K] (f16, row-major) @ W[N,K]^T (f16, N-major)
// One wave computes a 64x32 tile: 4 M-subtiles x 2 N-subtiles.
// k-loop unrolled x2 with two named fragment sets (no rotation copies);
// per-lane addresses resolved once, advanced by constant strides.
// NOTE: the last A-set prefetch overreads <=64B past the tile; all GEMM
// operands are interior workspace regions, so this stays in mapped memory
// and the fragments are never consumed.
// ---------------------------------------------------------------------------
__global__ void gemm_f16_f32(const _Float16* __restrict__ A,
                             const _Float16* __restrict__ W,
                             float* __restrict__ C,
                             int M, int N, int K) {
    int wave   = (blockIdx.x * blockDim.x + threadIdx.x) >> 5;
    int ntiles = N >> 5;                 // 32-wide column tiles
    int mt = wave / ntiles;
    int nt = wave % ntiles;
    if (mt * 64 >= M) return;
    int row0 = mt * 64, col0 = nt * 32;

    int l = threadIdx.x & 31;
    int lo16 = l & 15, hi = l >> 4;

    const _Float16* pa0 = A + (size_t)(row0 + lo16) * K + (hi << 3);
    const _Float16* pa1 = pa0 + (size_t)16 * K;
    const _Float16* pa2 = pa0 + (size_t)32 * K;
    const _Float16* pa3 = pa0 + (size_t)48 * K;
    const _Float16* pb0 = W + (size_t)(col0 + lo16) * K + (hi << 3);
    const _Float16* pb1 = pb0 + (size_t)16 * K;

    v8f acc[4][2] = {};
    v16h aA[4], b0A, b1A, aB[4], b0B, b1B;

    // prologue: set A <- k = 0
    b0A = load_b16(pb0);  b1A = load_b16(pb1);
    aA[0] = load_a16(pa0); aA[1] = load_a16(pa1);
    aA[2] = load_a16(pa2); aA[3] = load_a16(pa3);

    for (int k0 = 0; k0 < K; k0 += 64) {
        // set B <- k0+32 (in flight while set A computes)
        b0B = load_b16(pb0 + 32);  b1B = load_b16(pb1 + 32);
        aB[0] = load_a16(pa0 + 32); aB[1] = load_a16(pa1 + 32);
        aB[2] = load_a16(pa2 + 32); aB[3] = load_a16(pa3 + 32);

#pragma unroll
        for (int ms = 0; ms < 4; ++ms) {
            acc[ms][0] = wmma_f16(aA[ms], b0A, acc[ms][0]);
            acc[ms][1] = wmma_f16(aA[ms], b1A, acc[ms][1]);
        }

        pa0 += 64; pa1 += 64; pa2 += 64; pa3 += 64; pb0 += 64; pb1 += 64;

        // set A <- k0+64 (junk prefetch on the final iteration; unused)
        b0A = load_b16(pb0);  b1A = load_b16(pb1);
        aA[0] = load_a16(pa0); aA[1] = load_a16(pa1);
        aA[2] = load_a16(pa2); aA[3] = load_a16(pa3);

#pragma unroll
        for (int ms = 0; ms < 4; ++ms) {
            acc[ms][0] = wmma_f16(aB[ms], b0B, acc[ms][0]);
            acc[ms][1] = wmma_f16(aB[ms], b1B, acc[ms][1]);
        }
    }

    int mhalf = hi * 8;
#pragma unroll
    for (int ms = 0; ms < 4; ++ms)
#pragma unroll
        for (int ns = 0; ns < 2; ++ns)
#pragma unroll
            for (int i = 0; i < 8; ++i)
                C[(size_t)(row0 + ms * 16 + mhalf + i) * N + col0 + ns * 16 + lo16]
                    = acc[ms][ns][i];
}

// ---------------------------------------------------------------------------
// Per (t, unit) wave: RMSNorm + RoPE + gain on q/k heads, transpose-cast v.
// units: 0..15 = q heads, 16..19 = k heads, 20..23 = v heads.
// Qh:[16][T][64]  Kh:[4][T][64]  Vt:[4][64][T]   (all f16)
// ---------------------------------------------------------------------------
__global__ void qkv_post(const float* __restrict__ qkv,     // [T][1536]
                         const float* __restrict__ q_gain,  // [16]
                         _Float16* __restrict__ Qh,
                         _Float16* __restrict__ Kh,
                         _Float16* __restrict__ Vt) {
    int wid  = (blockIdx.x * blockDim.x + threadIdx.x) >> 5;
    int lane = threadIdx.x & 31;
    int t = wid / 24;
    int u = wid % 24;
    if (t >= T_LEN) return;

    if (u >= 20) {                                   // v heads: cast + transpose
        int h   = u - 20;
        int col = DIM + NKV * HD + h * HD;
        float v1 = qkv[(size_t)t * QKVN + col + lane];
        float v2 = qkv[(size_t)t * QKVN + col + lane + 32];
        Vt[(size_t)(h * HD + lane)      * T_LEN + t] = (_Float16)v1;
        Vt[(size_t)(h * HD + lane + 32) * T_LEN + t] = (_Float16)v2;
        return;
    }
    bool isq = (u < 16);
    int  h   = isq ? u : (u - 16);
    int  col = isq ? h * HD : DIM + h * HD;
    float v1 = qkv[(size_t)t * QKVN + col + lane];
    float v2 = qkv[(size_t)t * QKVN + col + lane + 32];

    // RMS over the 64 head dims (2 per lane, wave32 reduction)
    float ss = v1 * v1 + v2 * v2;
#pragma unroll
    for (int off = 16; off >= 1; off >>= 1) ss += __shfl_xor(ss, off, 32);
    float r = rsqrtf(ss * (1.0f / 64.0f) + 1.1920929e-7f);
    v1 *= r; v2 *= r;

    // RoPE: lane j pairs dims (j, j+32); inv_freq = BASE^(-2j/64)
    float inv_f = __expf(-((float)lane / 32.0f) * 9.210340371976184f); // ln(1e4)
    float f = (float)t * inv_f;
    float c = cosf(f), s = sinf(f);
    float o1 =  v1 * c + v2 * s;
    float o2 = -v1 * s + v2 * c;

    if (isq) {
        float g = q_gain[h];
        _Float16* dst = Qh + ((size_t)h * T_LEN + t) * HD;
        dst[lane]      = (_Float16)(o1 * g);
        dst[lane + 32] = (_Float16)(o2 * g);
    } else {
        _Float16* dst = Kh + ((size_t)h * T_LEN + t) * HD;
        dst[lane]      = (_Float16)o1;
        dst[lane + 32] = (_Float16)o2;
    }
}

// ---------------------------------------------------------------------------
// Causal flash attention. One wave = one head x one 16-row query tile.
// S = Q K^T via chained 16x16x32 WMMA (K=64); online softmax; P re-layout
// through a per-wave LDS tile; O += P V with Vt ([h][d][t], key-contiguous).
// K/V addresses are one running per-lane pointer each; sub-fragment offsets
// fold into the load IOFFSET. V fragments issue right after the S WMMAs so
// they overlap the softmax VALU work.
// ---------------------------------------------------------------------------
__global__ void flash_attn(const _Float16* __restrict__ Qh,
                           const _Float16* __restrict__ Kh,
                           const _Float16* __restrict__ Vt,
                           _Float16* __restrict__ Yh) {
    __shared__ _Float16 plds[4][16 * 32];            // per-wave P staging tile
    int wwave = threadIdx.x >> 5;
    int lane  = threadIdx.x & 31;
    int gwave = blockIdx.x * (blockDim.x >> 5) + wwave;
    int h  = gwave / (T_LEN / 16);
    int qt = gwave % (T_LEN / 16);
    int q0 = qt * 16;

    int lo16 = lane & 15, hi = lane >> 4;

    const _Float16* Qbase = Qh + ((size_t)h * T_LEN + q0) * HD;
    const _Float16* pQ = Qbase + (size_t)lo16 * HD + (hi << 3);
    v16h aq0 = load_a16(pQ);                          // d 0..31
    v16h aq1 = load_a16(pQ + 32);                     // d 32..63

    // running per-lane K/V fragment pointers
    const _Float16* pK = Kh + (size_t)(h >> 2) * T_LEN * HD
                            + (size_t)lo16 * HD + (hi << 3);
    const _Float16* pV = Vt + (size_t)(h >> 2) * HD * T_LEN
                            + (size_t)lo16 * T_LEN + (hi << 3);

    v8f acc[4] = {};
    float mrow[8], lrow[8];
#pragma unroll
    for (int i = 0; i < 8; ++i) { mrow[i] = -__builtin_inff(); lrow[i] = 0.0f; }

    _Float16* myLds = plds[wwave];
    const _Float16* pP = myLds + lo16 * 32 + (hi << 3);  // A-frag view of P

    for (int kb = 0; kb < q0 + 16; kb += 32) {
        // ---- S = scale * Q K^T for 2 key sub-tiles of 16 ----
        v8f s0 = {}, s1 = {};
        s0 = wmma_f16(aq0, load_b16(pK +    0), s0);     // keys kb..+15,  d 0..31
        s0 = wmma_f16(aq1, load_b16(pK +   32), s0);     // keys kb..+15,  d 32..63
        s1 = wmma_f16(aq0, load_b16(pK + 1024), s1);     // keys kb+16..+31, d 0..31
        s1 = wmma_f16(aq1, load_b16(pK + 1056), s1);     // keys kb+16..+31, d 32..63

        // ---- issue V fragment loads now; they overlap the softmax VALU ----
        v16h bv[4];
#pragma unroll
        for (int j = 0; j < 4; ++j)
            bv[j] = load_b16(pV + (size_t)j * 16 * T_LEN);

        pK += 32 * HD;
        pV += 32;

        // ---- scale + causal mask + online softmax update ----
#pragma unroll
        for (int i = 0; i < 8; ++i) {
            int q = q0 + hi * 8 + i;
            float e0 = s0[i] * 0.125f;
            float e1 = s1[i] * 0.125f;
            if (kb + lo16      > q) e0 = -__builtin_inff();
            if (kb + 16 + lo16 > q) e1 = -__builtin_inff();

            float mx = fmaxf(e0, e1);
#pragma unroll
            for (int off = 8; off >= 1; off >>= 1) mx = fmaxf(mx, __shfl_xor(mx, off, 32));
            float mnew  = fmaxf(mrow[i], mx);
            float alpha = __expf(mrow[i] - mnew);
            mrow[i] = mnew;

            float p0 = __expf(e0 - mnew);
            float p1 = __expf(e1 - mnew);
            float rs = p0 + p1;
#pragma unroll
            for (int off = 8; off >= 1; off >>= 1) rs += __shfl_xor(rs, off, 32);
            lrow[i] = lrow[i] * alpha + rs;

#pragma unroll
            for (int j = 0; j < 4; ++j) acc[j][i] *= alpha;

            // stage P (f16) in LDS: [row m][local key]
            int m = hi * 8 + i;
            myLds[m * 32 + lo16]      = (_Float16)p0;
            myLds[m * 32 + 16 + lo16] = (_Float16)p1;
        }
        asm volatile("s_wait_dscnt 0" ::: "memory");

        // ---- re-read P in A-fragment layout (16x32) ----
        v16h pa;
        {
            v8h plo = *(const v8h*)(pP);
            v8h phi = *(const v8h*)(pP + 16);
#pragma unroll
            for (int i = 0; i < 8; ++i) { pa[i] = plo[i]; pa[i + 8] = phi[i]; }
        }

        // ---- O += P V ----
#pragma unroll
        for (int j = 0; j < 4; ++j)
            acc[j] = wmma_f16(pa, bv[j], acc[j]);
    }

    // ---- normalize and write Yh[t][h*64+d] ----
#pragma unroll
    for (int i = 0; i < 8; ++i) {
        float inv = 1.0f / lrow[i];
        size_t row = (size_t)(q0 + hi * 8 + i);
        _Float16* dst = Yh + row * DIM + h * HD + lo16;
        dst[0]  = (_Float16)(acc[0][i] * inv);
        dst[16] = (_Float16)(acc[1][i] * inv);
        dst[32] = (_Float16)(acc[2][i] * inv);
        dst[48] = (_Float16)(acc[3][i] * inv);
    }
}

// ---------------------------------------------------------------------------
extern "C" void kernel_launch(void* const* d_in, const int* in_sizes, int n_in,
                              void* d_out, int out_size, void* d_ws, size_t ws_size,
                              hipStream_t stream) {
    const float* x      = (const float*)d_in[0];   // [T][DIM]
    const float* w_qkv  = (const float*)d_in[1];   // [QKVN][DIM]
    const float* w_proj = (const float*)d_in[2];   // [DIM][DIM]
    const float* q_gain = (const float*)d_in[3];   // [16]
    float* out = (float*)d_out;                    // [T][DIM]

    char* ws = (char*)d_ws;
    _Float16* Xh    = (_Float16*)ws; ws += (size_t)T_LEN * DIM  * 2;  //  8 MB
    _Float16* Wqkvh = (_Float16*)ws; ws += (size_t)QKVN  * DIM  * 2;  //  3 MB
    _Float16* Wph   = (_Float16*)ws; ws += (size_t)DIM   * DIM  * 2;  //  2 MB
    float*    qkv   = (float*)   ws; ws += (size_t)T_LEN * QKVN * 4;  // 24 MB
    _Float16* Qh    = (_Float16*)ws; ws += (size_t)NH  * T_LEN * HD * 2; // 8 MB
    _Float16* Kh    = (_Float16*)ws; ws += (size_t)NKV * T_LEN * HD * 2; // 2 MB
    _Float16* Vt    = (_Float16*)ws; ws += (size_t)NKV * HD * T_LEN * 2; // 2 MB
    _Float16* Yh    = (_Float16*)qkv;   // reuse qkv region (dead after qkv_post)

    // 1) f32 -> f16 casts
    cast_f32_to_f16<<<(T_LEN * DIM)  / 256, 256, 0, stream>>>(x,      Xh,    T_LEN * DIM);
    cast_f32_to_f16<<<(QKVN  * DIM)  / 256, 256, 0, stream>>>(w_qkv,  Wqkvh, QKVN  * DIM);
    cast_f32_to_f16<<<(DIM   * DIM)  / 256, 256, 0, stream>>>(w_proj, Wph,   DIM   * DIM);

    // 2) qkv = Xh @ Wqkv^T   (M=4096, N=1536, K=1024); waves = 64*48 = 3072
    gemm_f16_f32<<<(64 * 48) / 8, 256, 0, stream>>>(Xh, Wqkvh, qkv, T_LEN, QKVN, DIM);

    // 3) RMSNorm + RoPE + gain / V transpose: 4096*24 waves, 4 waves/block
    qkv_post<<<(T_LEN * 24) / 4, 128, 0, stream>>>(qkv, q_gain, Qh, Kh, Vt);

    // 4) causal flash attention: 16 heads * 256 q-tiles = 4096 waves
    flash_attn<<<(NH * (T_LEN / 16)) / 4, 128, 0, stream>>>(Qh, Kh, Vt, Yh);

    // 5) out = Yh @ w_proj^T  (M=4096, N=1024, K=1024); waves = 64*32 = 2048
    gemm_f16_f32<<<(64 * 32) / 8, 256, 0, stream>>>(Yh, Wph, out, T_LEN, DIM, DIM);
}
